// HungarianCurveMatcher_60361470378238
// MI455X (gfx1250) — matrix operand order repaired and checked by reading, not compiled, not run on hardware
//
#include <hip/hip_runtime.h>

typedef __attribute__((ext_vector_type(16))) _Float16 v16h;
typedef __attribute__((ext_vector_type(8)))  _Float16 v8h;
typedef __attribute__((ext_vector_type(4)))  _Float16 v4h;
typedef __attribute__((ext_vector_type(8)))  float    v8f;
typedef __attribute__((ext_vector_type(4)))  float    v4f;

#define B_      4
#define N_      100
#define M_      50
#define HW_     65536
#define NTILES  7
#define MTILES  4
#define NT_     (NTILES*16)   // 112 padded
#define MT_     (MTILES*16)   // 64 padded
#define KCHUNKS 32
#define KCHUNK  (HW_/KCHUNKS) // 2048
#define KSTEPS  (KCHUNK/32)   // 64
#define KSPLIT  8             // k-partitions per row in staging kernels

#define DMAX_   13.815510557964274f   // log((1-eps)/eps), eps=1e-6
#define EPS_    1e-6f
#define NEPSL_  -9.9999905e-7f        // log(1-eps)
#define LOG2E_  1.4426950408889634f
#define LN2_    0.6931471805599453f

// ---- workspace layout ----
#define OFF_DOTD  0
#define OFF_DOTP  (B_*NT_*MT_)
#define OFF_RSL   (2*B_*NT_*MT_)
#define OFF_RSP   (OFF_RSL + B_*NT_)
#define OFF_CST   (OFF_RSP + B_*NT_)
#define WS_FLOATS (OFF_CST + B_*MT_)              // 58496
#define ACC_BYTES ((size_t)WS_FLOATS*4)           // 233984 (16B aligned)
#define A_ELEMS   ((size_t)B_*N_*HW_)             // 26,214,400
#define T_ELEMS   ((size_t)B_*M_*HW_)             // 13,107,200
#define OFF_DH    ACC_BYTES
#define OFF_PH    (OFF_DH + A_ELEMS*2)
#define OFF_TH    (OFF_PH + A_ELEMS*2)
#define WS_STAGED (OFF_TH + T_ELEMS*2)            // ~131.3 MB

__global__ void hcm_zero_ws(float* __restrict__ ws) {
    int i = blockIdx.x * blockDim.x + threadIdx.x;
    if (i < WS_FLOATS) ws[i] = 0.0f;
}

// per-element conversion math (raw hw transcendentals, overflow-safe)
__device__ __forceinline__ void hcm_convert(float x, float& d, float& pm, float& l1m) {
    d = fminf(fmaxf(x, -DMAX_), DMAX_);                     // log(pm)-log(1-pm), clipped
    float ax  = fabsf(x);
    float emm = __builtin_amdgcn_exp2f(-ax * LOG2E_);       // e^{-|x|}, never overflows
    float den = 1.0f + emm;
    float rp  = __builtin_amdgcn_rcpf(den);                 // 1/(1+e^{-|x|})
    pm = (x >= 0.0f) ? rp : emm * rp;                       // sigmoid(x)
    pm = fminf(fmaxf(pm, EPS_), 1.0f - EPS_);
    // log(1-pm) = -softplus(x) = -(max(x,0) + ln(1+e^{-|x|}))
    float sp = fmaxf(x, 0.0f) + LN2_ * __builtin_amdgcn_logf(den);
    l1m = fminf(fmaxf(-sp, -DMAX_), NEPSL_);
}

// ---------- staging: pred_masks -> Dh,Ph (f16) + rowsums ----------
__global__ __launch_bounds__(256) void hcm_stage_a(
    const float* __restrict__ pred_masks, _Float16* __restrict__ Dh,
    _Float16* __restrict__ Ph, float* __restrict__ ws)
{
    __shared__ float red0[256], red1[256];
    const int kpart = blockIdx.x % KSPLIT;
    const int row   = blockIdx.x / KSPLIT;   // b*N_+n
    const int tid   = threadIdx.x;
    const size_t base = (size_t)row * HW_ + (size_t)kpart * (HW_ / KSPLIT);
    float sl = 0.0f, sp = 0.0f;
    const int ITER = HW_ / KSPLIT / (256 * 4); // 8
    for (int it = 0; it < ITER; ++it) {
        size_t off = base + (size_t)it * 1024 + tid * 4;
        v4f x4 = *(const v4f*)(pred_masks + off);
        v4h dv, pv;
#pragma unroll
        for (int j = 0; j < 4; ++j) {
            float d, pm, l1m;
            hcm_convert(x4[j], d, pm, l1m);
            sl += l1m; sp += pm;
            dv[j] = (_Float16)d; pv[j] = (_Float16)pm;
        }
        *(v4h*)(Dh + off) = dv;
        *(v4h*)(Ph + off) = pv;
    }
    red0[tid] = sl; red1[tid] = sp;
    __syncthreads();
    for (int s = 128; s > 0; s >>= 1) {
        if (tid < s) { red0[tid] += red0[tid + s]; red1[tid] += red1[tid + s]; }
        __syncthreads();
    }
    if (tid == 0) {
        const int b = row / N_, n = row % N_;
        unsafeAtomicAdd(&ws[OFF_RSL + b * NT_ + n], red0[0]);
        unsafeAtomicAdd(&ws[OFF_RSP + b * NT_ + n], red1[0]);
    }
}

// ---------- staging: tgt_masks -> Th (f16) + colsums ----------
__global__ __launch_bounds__(256) void hcm_stage_b(
    const float* __restrict__ tgt_masks, _Float16* __restrict__ Th,
    float* __restrict__ ws)
{
    __shared__ float red0[256];
    const int kpart = blockIdx.x % KSPLIT;
    const int row   = blockIdx.x / KSPLIT;   // b*M_+m
    const int tid   = threadIdx.x;
    const size_t base = (size_t)row * HW_ + (size_t)kpart * (HW_ / KSPLIT);
    float st = 0.0f;
    const int ITER = HW_ / KSPLIT / (256 * 4);
    for (int it = 0; it < ITER; ++it) {
        size_t off = base + (size_t)it * 1024 + tid * 4;
        v4f t4 = *(const v4f*)(tgt_masks + off);
        v4h tv;
#pragma unroll
        for (int j = 0; j < 4; ++j) { st += t4[j]; tv[j] = (_Float16)t4[j]; }
        *(v4h*)(Th + off) = tv;
    }
    red0[tid] = st;
    __syncthreads();
    for (int s = 128; s > 0; s >>= 1) {
        if (tid < s) red0[tid] += red0[tid + s];
        __syncthreads();
    }
    if (tid == 0) {
        const int b = row / M_, m = row % M_;
        unsafeAtomicAdd(&ws[OFF_CST + b * MT_ + m], red0[0]);
    }
}

// ---------- staged GEMM: branch-free f16 loads + WMMA ----------
// Out-of-range rows/cols are CLAMPED to valid addresses: the garbage they
// produce lands only in padded output rows (>=N_) / cols (>=M_) that the
// finalize kernel never reads. No exec masking, no zero-fill in the loop.
union V16U { v16h v; v8h h[2]; };

__global__ __launch_bounds__(32) void hcm_gemm_staged(
    const _Float16* __restrict__ Dh, const _Float16* __restrict__ Ph,
    const _Float16* __restrict__ Th, float* __restrict__ ws)
{
    const int lane = threadIdx.x;
    const int kc = blockIdx.x % KCHUNKS;
    const int nt = (blockIdx.x / KCHUNKS) % NTILES;
    const int b  = blockIdx.x / (KCHUNKS * NTILES);
    const int r  = lane & 15;
    const int hi = lane >> 4;
    const int row  = nt * 16 + r;
    const int rowC = row < N_ ? row : (N_ - 1);          // clamp (see note)
    const size_t arow = (size_t)(b * N_ + rowC) * HW_;

    // hoisted per-lane base pointers
    const int k0 = kc * KCHUNK;
    const _Float16* pD = Dh + arow + k0 + 8 * hi;        // A halves 0..7 here, 8..15 at +16
    const _Float16* pP = Ph + arow + k0 + 8 * hi;
    const _Float16* pT[MTILES];
#pragma unroll
    for (int mt = 0; mt < MTILES; ++mt) {
        const int m  = mt * 16 + r;
        const int mC = m < M_ ? m : (M_ - 1);            // clamp (see note)
        pT[mt] = Th + (size_t)(b * M_ + mC) * HW_ + k0 + 8 * hi;
    }

    v8f accD[MTILES], accP[MTILES];
    {
        v8f z = {};
#pragma unroll
        for (int mt = 0; mt < MTILES; ++mt) { accD[mt] = z; accP[mt] = z; }
    }

    for (int ks = 0; ks < KSTEPS; ++ks) {
        const int kb = ks * 32;
        V16U aD, aP;
        aD.h[0] = *(const v8h*)(pD + kb);
        aD.h[1] = *(const v8h*)(pD + kb + 16);
        aP.h[0] = *(const v8h*)(pP + kb);
        aP.h[1] = *(const v8h*)(pP + kb + 16);
#pragma unroll
        for (int mt = 0; mt < MTILES; ++mt) {
            V16U bf;
            bf.h[0] = *(const v8h*)(pT[mt] + kb);
            bf.h[1] = *(const v8h*)(pT[mt] + kb + 16);
            accD[mt] = __builtin_amdgcn_wmma_f32_16x16x32_f16(
                false, aD.v, false, bf.v, (short)0, accD[mt], false, false);
            accP[mt] = __builtin_amdgcn_wmma_f32_16x16x32_f16(
                false, aP.v, false, bf.v, (short)0, accP[mt], false, false);
        }
    }

    float* dotD = ws + OFF_DOTD;
    float* dotP = ws + OFF_DOTP;
#pragma unroll
    for (int mt = 0; mt < MTILES; ++mt) {
#pragma unroll
        for (int i = 0; i < 8; ++i) {
            const int on = nt * 16 + i + 8 * hi;
            const int om = mt * 16 + r;
            const size_t o = ((size_t)b * NT_ + on) * MT_ + om;
            unsafeAtomicAdd(&dotD[o], accD[mt][i]);
            unsafeAtomicAdd(&dotP[o], accP[mt][i]);
        }
    }
}

// ---------- fused fallback (small ws): convert on the fly ----------
// Same clamping trick: clamped lanes' sums land in padded rs/cst slots.
__global__ __launch_bounds__(32) void hcm_gemm_fused(
    const float* __restrict__ pred_masks, const float* __restrict__ tgt_masks,
    float* __restrict__ ws)
{
    const int lane = threadIdx.x;
    const int kc = blockIdx.x % KCHUNKS;
    const int nt = (blockIdx.x / KCHUNKS) % NTILES;
    const int b  = blockIdx.x / (KCHUNKS * NTILES);
    const int r  = lane & 15;
    const int hi = lane >> 4;
    const int row  = nt * 16 + r;
    const int rowC = row < N_ ? row : (N_ - 1);
    const int k0 = kc * KCHUNK;
    const float* pA = pred_masks + (size_t)(b * N_ + rowC) * HW_ + k0 + 8 * hi;
    const float* pT[MTILES];
#pragma unroll
    for (int mt = 0; mt < MTILES; ++mt) {
        const int m  = mt * 16 + r;
        const int mC = m < M_ ? m : (M_ - 1);
        pT[mt] = tgt_masks + (size_t)(b * M_ + mC) * HW_ + k0 + 8 * hi;
    }

    v8f accD[MTILES], accP[MTILES];
    {
        v8f z = {};
#pragma unroll
        for (int mt = 0; mt < MTILES; ++mt) { accD[mt] = z; accP[mt] = z; }
    }
    float sl1m = 0.0f, spm = 0.0f;
    float stm[MTILES] = {0.0f, 0.0f, 0.0f, 0.0f};

    for (int ks = 0; ks < KSTEPS; ++ks) {
        const int kb = ks * 32;
        v4f av[4];
        av[0] = *(const v4f*)(pA + kb);
        av[1] = *(const v4f*)(pA + kb + 4);
        av[2] = *(const v4f*)(pA + kb + 16);
        av[3] = *(const v4f*)(pA + kb + 20);
        v16h aD, aP;
#pragma unroll
        for (int j = 0; j < 16; ++j) {
            float d, pm, l1m;
            hcm_convert(av[j >> 2][j & 3], d, pm, l1m);
            sl1m += l1m; spm += pm;
            aD[j] = (_Float16)d; aP[j] = (_Float16)pm;
        }
#pragma unroll
        for (int mt = 0; mt < MTILES; ++mt) {
            v4f tv[4];
            tv[0] = *(const v4f*)(pT[mt] + kb);
            tv[1] = *(const v4f*)(pT[mt] + kb + 4);
            tv[2] = *(const v4f*)(pT[mt] + kb + 16);
            tv[3] = *(const v4f*)(pT[mt] + kb + 20);
            v16h bf;
#pragma unroll
            for (int j = 0; j < 16; ++j) bf[j] = (_Float16)tv[j >> 2][j & 3];
            if (nt == 0) { // wave-uniform branch
                float ssum = 0.0f;
#pragma unroll
                for (int j = 0; j < 16; ++j) ssum += tv[j >> 2][j & 3];
                stm[mt] += ssum;
            }
            accD[mt] = __builtin_amdgcn_wmma_f32_16x16x32_f16(
                false, aD, false, bf, (short)0, accD[mt], false, false);
            accP[mt] = __builtin_amdgcn_wmma_f32_16x16x32_f16(
                false, aP, false, bf, (short)0, accP[mt], false, false);
        }
    }

    float* dotD = ws + OFF_DOTD;
    float* dotP = ws + OFF_DOTP;
#pragma unroll
    for (int mt = 0; mt < MTILES; ++mt) {
#pragma unroll
        for (int i = 0; i < 8; ++i) {
            const int on = nt * 16 + i + 8 * hi;
            const int om = mt * 16 + r;
            const size_t o = ((size_t)b * NT_ + on) * MT_ + om;
            unsafeAtomicAdd(&dotD[o], accD[mt][i]);
            unsafeAtomicAdd(&dotP[o], accP[mt][i]);
        }
    }
    unsafeAtomicAdd(&ws[OFF_RSL + b * NT_ + row], sl1m);
    unsafeAtomicAdd(&ws[OFF_RSP + b * NT_ + row], spm);
    if (nt == 0) {
#pragma unroll
        for (int mt = 0; mt < MTILES; ++mt)
            unsafeAtomicAdd(&ws[OFF_CST + b * MT_ + mt * 16 + r], stm[mt]);
    }
}

// ---------- finalize ----------
__global__ void hcm_finalize(
    const float* __restrict__ pred_logits, const float* __restrict__ pred_style,
    const int* __restrict__ styles, const float* __restrict__ ws,
    float* __restrict__ out)
{
    int idx = blockIdx.x * blockDim.x + threadIdx.x;
    if (idx >= B_ * N_ * M_) return;
    const int m = idx % M_;
    const int n = (idx / M_) % N_;
    const int b = idx / (M_ * N_);

    const size_t o = ((size_t)b * NT_ + n) * MT_ + m;
    const float dD  = ws[OFF_DOTD + o];
    const float dP  = ws[OFF_DOTP + o];
    const float rs1 = ws[OFF_RSL + b * NT_ + n];
    const float spm = ws[OFF_RSP + b * NT_ + n];
    const float stm = ws[OFF_CST + b * MT_ + m];

    const float cost_mask = -(dD + rs1) * (1.0f / (float)HW_);
    const float cost_dice = 1.0f - (2.0f * dP + 1.0f) / (spm + stm + 1.0f);

    const float x0 = pred_logits[(b * N_ + n) * 2 + 0];
    const float x1 = pred_logits[(b * N_ + n) * 2 + 1];
    const float p1 = 1.0f / (1.0f + __expf(x0 - x1));

    const float* st = pred_style + (b * N_ + n) * 4;
    const float mx = fmaxf(fmaxf(st[0], st[1]), fmaxf(st[2], st[3]));
    const float e0 = __expf(st[0] - mx), e1 = __expf(st[1] - mx);
    const float e2 = __expf(st[2] - mx), e3 = __expf(st[3] - mx);
    int sid = styles[b * M_ + m];
    sid = sid < 0 ? 0 : (sid > 3 ? 3 : sid);
    const float es = (sid == 0) ? e0 : (sid == 1) ? e1 : (sid == 2) ? e2 : e3;
    const float sp = es / (e0 + e1 + e2 + e3);

    float cost = 2.0f * (-p1) + 5.0f * cost_mask + 5.0f * cost_dice - sp;
    if (__builtin_isnan(cost))      cost = 10000.0f;
    else if (__builtin_isinf(cost)) cost = (cost > 0.0f) ? 10000.0f : -10000.0f;
    out[idx] = cost;
}

extern "C" void kernel_launch(void* const* d_in, const int* in_sizes, int n_in,
                              void* d_out, int out_size, void* d_ws, size_t ws_size,
                              hipStream_t stream) {
    const float* pred_logits = (const float*)d_in[0];
    const float* pred_masks  = (const float*)d_in[1];
    const float* pred_style  = (const float*)d_in[2];
    const float* tgt_masks   = (const float*)d_in[3];
    const int*   styles      = (const int*)d_in[4];
    float* ws  = (float*)d_ws;
    float* out = (float*)d_out;

    hcm_zero_ws<<<(WS_FLOATS + 255) / 256, 256, 0, stream>>>(ws);

    if (ws_size >= WS_STAGED) {
        _Float16* Dh = (_Float16*)((char*)d_ws + OFF_DH);
        _Float16* Ph = (_Float16*)((char*)d_ws + OFF_PH);
        _Float16* Th = (_Float16*)((char*)d_ws + OFF_TH);
        hcm_stage_a<<<B_ * N_ * KSPLIT, 256, 0, stream>>>(pred_masks, Dh, Ph, ws);
        hcm_stage_b<<<B_ * M_ * KSPLIT, 256, 0, stream>>>(tgt_masks, Th, ws);
        hcm_gemm_staged<<<B_ * NTILES * KCHUNKS, 32, 0, stream>>>(Dh, Ph, Th, ws);
    } else {
        hcm_gemm_fused<<<B_ * NTILES * KCHUNKS, 32, 0, stream>>>(pred_masks, tgt_masks, ws);
    }

    hcm_finalize<<<(B_ * N_ * M_ + 255) / 256, 256, 0, stream>>>(
        pred_logits, pred_style, styles, ws, out);
}
